// HGRNBitMLP_455266534017
// MI455X (gfx1250) — 8-bit matrix operand order repaired and checked by reading, compiled, not run on hardware
//
#include <hip/hip_runtime.h>

typedef int v8i __attribute__((ext_vector_type(8)));

#define EPS_NORM 1e-8f
#define EPS_Q    1e-5f

__global__ __launch_bounds__(256) void hgrn_absmean_partial(
    const float* __restrict__ W, long n, float* __restrict__ partial) {
  __shared__ float red[256];
  long i0 = (long)blockIdx.x * blockDim.x + threadIdx.x;
  long stride = (long)gridDim.x * blockDim.x;
  float s = 0.f;
  for (long i = i0; i < n; i += stride) s += fabsf(W[i]);
  red[threadIdx.x] = s; __syncthreads();
  for (int t = 128; t > 0; t >>= 1) {
    if ((int)threadIdx.x < t) red[threadIdx.x] += red[threadIdx.x + t];
    __syncthreads();
  }
  if (threadIdx.x == 0) partial[blockIdx.x] = red[0];
}

__global__ __launch_bounds__(256) void hgrn_finalize_scales(
    const float* __restrict__ pg, int ng, long cntg,
    const float* __restrict__ pd, int nd, long cntd,
    float* __restrict__ scales) {
  __shared__ float red[256];
  int tid = threadIdx.x;
  float s = 0.f;
  for (int i = tid; i < ng; i += 256) s += pg[i];
  red[tid] = s; __syncthreads();
  for (int t = 128; t > 0; t >>= 1) { if (tid < t) red[tid] += red[tid + t]; __syncthreads(); }
  if (tid == 0) {
    float c = fmaxf(red[0] / (float)cntg, EPS_Q);
    scales[0] = c; scales[2] = 1.f / c;
  }
  __syncthreads();
  s = 0.f;
  for (int i = tid; i < nd; i += 256) s += pd[i];
  red[tid] = s; __syncthreads();
  for (int t = 128; t > 0; t >>= 1) { if (tid < t) red[tid] += red[tid + t]; __syncthreads(); }
  if (tid == 0) {
    float c = fmaxf(red[0] / (float)cntd, EPS_Q);
    scales[1] = c; scales[3] = 1.f / c;
  }
}

__global__ __launch_bounds__(256) void hgrn_quant_weight(
    const float* __restrict__ W, long n, const float* __restrict__ scales, int sidx,
    signed char* __restrict__ Q) {
  float qs = scales[sidx];
  long i0 = (long)blockIdx.x * blockDim.x + threadIdx.x;
  long stride = (long)gridDim.x * blockDim.x;
  for (long i = i0; i < n; i += stride) {
    float q = rintf(W[i] * qs);
    q = fminf(1.f, fmaxf(-1.f, q));
    Q[i] = (signed char)(int)q;
  }
}

__global__ __launch_bounds__(256) void hgrn_act_quant(
    const float* __restrict__ X, const float* __restrict__ NW,
    signed char* __restrict__ Q, float* __restrict__ S, int Hdim) {
  __shared__ float red[256];
  int row = blockIdx.x;
  int tid = threadIdx.x;
  const float* x = X + (long)row * Hdim;

  float ss = 0.f;
  for (int i = tid; i < Hdim; i += 256) { float v = x[i]; ss += v * v; }
  red[tid] = ss; __syncthreads();
  for (int t = 128; t > 0; t >>= 1) { if (tid < t) red[tid] += red[tid + t]; __syncthreads(); }
  float rms = rsqrtf(red[0] / (float)Hdim + EPS_NORM);
  __syncthreads();

  float mx = 0.f;
  for (int i = tid; i < Hdim; i += 256) {
    float v = x[i] * rms * NW[i];
    mx = fmaxf(mx, fabsf(v));
  }
  red[tid] = mx; __syncthreads();
  for (int t = 128; t > 0; t >>= 1) { if (tid < t) red[tid] = fmaxf(red[tid], red[tid + t]); __syncthreads(); }
  float amax = fmaxf(red[0], EPS_Q);
  float scale = 127.f / amax;

  signed char* q = Q + (long)row * Hdim;
  for (int i = tid; i < Hdim; i += 256) {
    float v = x[i] * rms * NW[i] * scale;
    float r = fminf(127.f, fmaxf(-128.f, rintf(v)));
    q[i] = (signed char)(int)r;
  }
  if (tid == 0) S[row] = amax / 127.f;
}

__device__ __forceinline__ v8i hgrn_load_a(
    const signed char* __restrict__ A, int lda, int row0, int k0, int lane) {
  int r = row0 + (lane & 15);
  int half = lane >> 4;
  const signed char* p = A + (long)r * lda + k0 + half * 16;
  v8i a;
  int2 q;
  q = *(const int2*)(p +  0); a[0] = q.x; a[1] = q.y;
  q = *(const int2*)(p +  8); a[2] = q.x; a[3] = q.y;
  q = *(const int2*)(p + 32); a[4] = q.x; a[5] = q.y;
  q = *(const int2*)(p + 40); a[6] = q.x; a[7] = q.y;
  return a;
}

__device__ __forceinline__ v8i hgrn_load_b(
    const signed char* __restrict__ B, int ldb, int col0, int k0, int lane) {
  int c = col0 + (lane & 15);
  int half = lane >> 4;
  const signed char* p = B + (long)c * ldb + k0 + half * 16;
  v8i b;
  int4 q;
  q = *(const int4*)(p +  0); b[0] = q.x; b[1] = q.y; b[2] = q.z; b[3] = q.w;
  q = *(const int4*)(p + 32); b[4] = q.x; b[5] = q.y; b[6] = q.z; b[7] = q.w;
  return b;
}

__device__ __forceinline__ v8i hgrn_wmma(v8i a, v8i b, v8i c) {
  return __builtin_amdgcn_wmma_i32_16x16x64_iu8(true, a, true, b, c, false, false);
}

__global__ __launch_bounds__(256) void hgrn_gemm1_swiglu(
    const signed char* __restrict__ Aq, const float* __restrict__ Asc,
    const signed char* __restrict__ Wq, const float* __restrict__ scales,
    float* __restrict__ Hout, int K, int I) {
  const int lane = threadIdx.x & 31;
  const int wave = threadIdx.x >> 5;
  const int m0 = blockIdx.y * 64 + (wave >> 2) * 32;
  const int n0 = blockIdx.x * 128 + (wave & 3) * 32;
  const float wsc = scales[0];

  const v8i vz = {0, 0, 0, 0, 0, 0, 0, 0};
  v8i ag[2][2], au[2][2];
  for (int i = 0; i < 2; i++)
    for (int j = 0; j < 2; j++) { ag[i][j] = vz; au[i][j] = vz; }

  for (int k0 = 0; k0 < K; k0 += 64) {
    v8i a0  = hgrn_load_a(Aq, K, m0,      k0, lane);
    v8i a1  = hgrn_load_a(Aq, K, m0 + 16, k0, lane);
    v8i bg0 = hgrn_load_b(Wq, K, n0,          k0, lane);
    v8i bg1 = hgrn_load_b(Wq, K, n0 + 16,     k0, lane);
    v8i bu0 = hgrn_load_b(Wq, K, I + n0,      k0, lane);
    v8i bu1 = hgrn_load_b(Wq, K, I + n0 + 16, k0, lane);
    ag[0][0] = hgrn_wmma(a0, bg0, ag[0][0]);
    ag[0][1] = hgrn_wmma(a0, bg1, ag[0][1]);
    ag[1][0] = hgrn_wmma(a1, bg0, ag[1][0]);
    ag[1][1] = hgrn_wmma(a1, bg1, ag[1][1]);
    au[0][0] = hgrn_wmma(a0, bu0, au[0][0]);
    au[0][1] = hgrn_wmma(a0, bu1, au[0][1]);
    au[1][0] = hgrn_wmma(a1, bu0, au[1][0]);
    au[1][1] = hgrn_wmma(a1, bu1, au[1][1]);
  }

  const int half = lane >> 4;
  const int nn = lane & 15;
  for (int mi = 0; mi < 2; mi++) {
    for (int ni = 0; ni < 2; ni++) {
      int nb = n0 + ni * 16 + nn;
      for (int r = 0; r < 8; r++) {
        int m = m0 + mi * 16 + r + 8 * half;
        float sx = Asc[m] * wsc;
        float g = (float)ag[mi][ni][r] * sx;
        float u = (float)au[mi][ni][r] * sx;
        float silu = g / (1.f + expf(-g));
        Hout[(long)m * I + nb] = silu * u;
      }
    }
  }
}

__global__ __launch_bounds__(256) void hgrn_gemm2(
    const signed char* __restrict__ Aq, const float* __restrict__ Asc,
    const signed char* __restrict__ Wq, const float* __restrict__ scales,
    float* __restrict__ Out, int K, int N) {
  const int lane = threadIdx.x & 31;
  const int wave = threadIdx.x >> 5;
  const int m0 = blockIdx.y * 64 + (wave >> 2) * 32;
  const int n0 = blockIdx.x * 256 + (wave & 3) * 64;
  const float wsc = scales[1];

  const v8i vz = {0, 0, 0, 0, 0, 0, 0, 0};
  v8i acc[2][4];
  for (int i = 0; i < 2; i++)
    for (int j = 0; j < 4; j++) acc[i][j] = vz;

  for (int k0 = 0; k0 < K; k0 += 64) {
    v8i a0 = hgrn_load_a(Aq, K, m0,      k0, lane);
    v8i a1 = hgrn_load_a(Aq, K, m0 + 16, k0, lane);
    v8i b0 = hgrn_load_b(Wq, K, n0,      k0, lane);
    v8i b1 = hgrn_load_b(Wq, K, n0 + 16, k0, lane);
    v8i b2 = hgrn_load_b(Wq, K, n0 + 32, k0, lane);
    v8i b3 = hgrn_load_b(Wq, K, n0 + 48, k0, lane);
    acc[0][0] = hgrn_wmma(a0, b0, acc[0][0]);
    acc[0][1] = hgrn_wmma(a0, b1, acc[0][1]);
    acc[0][2] = hgrn_wmma(a0, b2, acc[0][2]);
    acc[0][3] = hgrn_wmma(a0, b3, acc[0][3]);
    acc[1][0] = hgrn_wmma(a1, b0, acc[1][0]);
    acc[1][1] = hgrn_wmma(a1, b1, acc[1][1]);
    acc[1][2] = hgrn_wmma(a1, b2, acc[1][2]);
    acc[1][3] = hgrn_wmma(a1, b3, acc[1][3]);
  }

  const int half = lane >> 4;
  const int nn = lane & 15;
  for (int mi = 0; mi < 2; mi++) {
    for (int ni = 0; ni < 4; ni++) {
      int nb = n0 + ni * 16 + nn;
      for (int r = 0; r < 8; r++) {
        int m = m0 + mi * 16 + r + 8 * half;
        Out[(long)m * N + nb] = (float)acc[mi][ni][r] * Asc[m] * wsc;
      }
    }
  }
}

extern "C" void kernel_launch(void* const* d_in, const int* in_sizes, int n_in,
                              void* d_out, int out_size, void* d_ws, size_t ws_size,
                              hipStream_t stream) {
  const float* x         = (const float*)d_in[0];
  const float* w_gate    = (const float*)d_in[1];
  const float* norm_gate = (const float*)d_in[2];
  const float* w_down    = (const float*)d_in[3];
  const float* norm_down = (const float*)d_in[4];
  float* out = (float*)d_out;

  const int H = in_sizes[2];
  const int I = in_sizes[4];
  const int M = in_sizes[0] / H;
  const long n_wg = (long)in_sizes[1];
  const long n_wd = (long)in_sizes[3];

  char* ws = (char*)d_ws;
  size_t off = 0;
  auto take = [&](size_t bytes) -> char* {
    char* p = ws + off;
    off += (bytes + 255) & ~(size_t)255;
    return p;
  };
  signed char* wq_gate = (signed char*)take((size_t)n_wg);
  signed char* wq_down = (signed char*)take((size_t)n_wd);
  signed char* xq      = (signed char*)take((size_t)M * H);
  signed char* hq      = (signed char*)take((size_t)M * I);
  float*       hbuf    = (float*)take((size_t)M * I * sizeof(float));
  float*       xsc     = (float*)take((size_t)M * sizeof(float));
  float*       hsc     = (float*)take((size_t)M * sizeof(float));
  float*       partG   = (float*)take(512 * sizeof(float));
  float*       partD   = (float*)take(512 * sizeof(float));
  float*       scales  = (float*)take(4 * sizeof(float));
  (void)ws_size;

  hgrn_absmean_partial<<<512, 256, 0, stream>>>(w_gate, n_wg, partG);
  hgrn_absmean_partial<<<512, 256, 0, stream>>>(w_down, n_wd, partD);
  hgrn_finalize_scales<<<1, 256, 0, stream>>>(partG, 512, n_wg, partD, 512, n_wd, scales);
  hgrn_quant_weight<<<2048, 256, 0, stream>>>(w_gate, n_wg, scales, 2, wq_gate);
  hgrn_quant_weight<<<1024, 256, 0, stream>>>(w_down, n_wd, scales, 3, wq_down);

  hgrn_act_quant<<<M, 256, 0, stream>>>(x, norm_gate, xq, xsc, H);

  {
    dim3 grid(I / 128, M / 64);
    hgrn_gemm1_swiglu<<<grid, 256, 0, stream>>>(xq, xsc, wq_gate, scales, hbuf, H, I);
  }

  hgrn_act_quant<<<M, 256, 0, stream>>>(hbuf, norm_down, hq, hsc, I);

  {
    dim3 grid(H / 256, M / 64);
    hgrn_gemm2<<<grid, 256, 0, stream>>>(hq, hsc, wq_down, scales, out, I, H);
  }
}
